// DynamicEncoder_23227183137097
// MI455X (gfx1250) — matrix operand order repaired and checked by reading, not compiled, not run on hardware
//
#include <hip/hip_runtime.h>
#include <hip/hip_bf16.h>
#include <stdint.h>
#include <stddef.h>

// ---------------------------------------------------------------------------
// Problem constants (match reference)
#define F_IN   128
#define HID    256
#define RELS   4
#define HEADS  4
#define HEADC  64

// ---------------------------------------------------------------------------
// CDNA5 WMMA types
typedef __attribute__((ext_vector_type(16))) __bf16 v16bf;
typedef __attribute__((ext_vector_type(8)))  float  v8f;

// CDNA5 async global->LDS (ASYNCcnt) support detection.
// Probe-learned signature: (int4 AS1*, int4 AS3*, imm offset, imm cpol).
#define AS1 __attribute__((address_space(1)))
#define AS3 __attribute__((address_space(3)))
typedef int v4i_vs __attribute__((vector_size(4 * sizeof(int))));
#if defined(__has_builtin)
#if __has_builtin(__builtin_amdgcn_global_load_async_to_lds_b128)
#define HAVE_ASYNC_LDS 1
#endif
#if __has_builtin(__builtin_amdgcn_s_wait_asynccnt)
#define HAVE_WAIT_ASYNC_BUILTIN 1
#endif
#endif

__device__ __forceinline__ void wait_asynccnt0() {
#if defined(HAVE_WAIT_ASYNC_BUILTIN)
  __builtin_amdgcn_s_wait_asynccnt(0);
#else
  asm volatile("s_wait_asynccnt 0x0" ::: "memory");
#endif
}

__device__ __forceinline__ uint16_t f32_to_bf16_rne(float f) {
  uint32_t u = __float_as_uint(f);
  u += 0x7FFFu + ((u >> 16) & 1u);   // round-to-nearest-even
  return (uint16_t)(u >> 16);
}

__device__ __forceinline__ unsigned ford_enc(float f) {
  unsigned u = __float_as_uint(f);
  return (u & 0x80000000u) ? ~u : (u | 0x80000000u);
}
__device__ __forceinline__ float ford_dec(unsigned e) {
  unsigned u = (e & 0x80000000u) ? (e & 0x7FFFFFFFu) : ~e;
  return __uint_as_float(u);
}
__device__ __forceinline__ float sigmoidf_(float x) {
  return 1.0f / (1.0f + expf(-x));
}

// ---------------------------------------------------------------------------
// bf16 WMMA GEMM:  Y[M,Nn] = act( A[M,K](f32) * B[K,Nn] + bias )
// B supplied pre-transposed + pre-converted:  BT[Nn][K] bf16.
// Block = 256 threads = 8 waves (wave32). Block tile 128x64, K-step 32.
// Wave w computes rows [16w,16w+16) x 64 cols via 4 f32 accumulators.
// B tile is staged with GLOBAL_LOAD_ASYNC_TO_LDS_B128 (ASYNCcnt) when the
// toolchain exposes it; the fetch overlaps the A-tile f32->bf16 VALU staging.
#define BM 128
#define BN 64
#define BK 32
#define APAD 34   // halfwords per A-tile row in LDS (bank-conflict-free dword reads)
#define BPAD 40   // halfwords per B-tile row in LDS (80B rows => 16B-aligned b128)

__global__ __launch_bounds__(256)
void gemm_bf16_wmma(const float* __restrict__ A, int lda,
                    const uint16_t* __restrict__ BT,      // [Nn][K] bf16
                    const float* __restrict__ bias,       // [Nn]
                    float* __restrict__ Y, int ldy,
                    int M, int K, int Nn, int relu) {
  __shared__ uint16_t sA[BM * APAD];
  __shared__ uint16_t sB[BN * BPAD];

  const int tid  = threadIdx.x;
  const int wave = tid >> 5;
  const int lane = tid & 31;
  const int lo16 = lane & 15;
  const int hi   = lane >> 4;
  const int m0   = blockIdx.x * BM;
  const int n0   = blockIdx.y * BN;

  v8f acc[4];
  const v8f vzero = {0.f, 0.f, 0.f, 0.f, 0.f, 0.f, 0.f, 0.f};
#pragma unroll
  for (int t = 0; t < 4; ++t) acc[t] = vzero;

  // B-tile staging coordinates (one b128 per thread: 64 rows x 4 segments)
  const int brow = tid >> 2;            // 0..63 (output column n0+brow)
  const int bseg = (tid & 3) << 3;      // halfword offset 0,8,16,24

  for (int k0 = 0; k0 < K; k0 += BK) {
    // ---- B tile: 64 cols x 32 bf16, direct global->LDS (no VGPR round trip)
#if defined(HAVE_ASYNC_LDS)
    __builtin_amdgcn_global_load_async_to_lds_b128(
        (AS1 v4i_vs*)(BT + (size_t)(n0 + brow) * K + k0 + bseg),
        (AS3 v4i_vs*)&sB[brow * BPAD + bseg], 0, 0);
#else
    {
      uint4 v = *(const uint4*)(BT + (size_t)(n0 + brow) * K + k0 + bseg);
      *(uint4*)&sB[brow * BPAD + bseg] = v;
    }
#endif
    // ---- stage A tile: 128 x 32 f32 -> bf16 (1024 float4 loads, 4/thread)
#pragma unroll
    for (int i = 0; i < 4; ++i) {
      int idx  = tid + (i << 8);          // 0..1023
      int row  = idx >> 3;
      int col4 = (idx & 7) << 2;
      int gm   = m0 + row;
      float4 v = make_float4(0.f, 0.f, 0.f, 0.f);
      if (gm < M) v = *(const float4*)(A + (size_t)gm * lda + k0 + col4);
      uint16_t* d = &sA[row * APAD + col4];
      d[0] = f32_to_bf16_rne(v.x); d[1] = f32_to_bf16_rne(v.y);
      d[2] = f32_to_bf16_rne(v.z); d[3] = f32_to_bf16_rne(v.w);
    }
    // prefetch next A K-slab (emits global_prefetch_b8)
    if (k0 + BK < K) {
      int prow = m0 + (tid & 127);
      if (prow < M) __builtin_prefetch(A + (size_t)prow * lda + k0 + BK, 0, 1);
    }
#if defined(HAVE_ASYNC_LDS)
    wait_asynccnt0();                    // own async LDS writes landed
#endif
    __syncthreads();                     // all waves' tiles visible

    // ---- A fragment (16x32 bf16, documented 16-bit A layout):
    // lane lo16 = M-row; lo lanes hold K 0..7 & 16..23, hi lanes +8.
    union { v16bf v; uint32_t u[8]; } af;
    {
      const uint16_t* ap = &sA[((wave << 4) + lo16) * APAD];
      const int khalf = hi << 3;          // 0 or 8
#pragma unroll
      for (int j = 0; j < 8; ++j) {
        int kk = ((j < 4) ? (j << 1) : ((j << 1) + 8)) + khalf;
        af.u[j] = *(const uint32_t*)(ap + kk);
      }
    }
    // ---- 4 B fragments (32x16 bf16: lane lo16 = N-col; lo lanes K 0..7 & 16..23,
    // hi lanes +8) + WMMA
#pragma unroll
    for (int t = 0; t < 4; ++t) {
      union { v16bf v; uint32_t u[8]; } bf;
      const uint16_t* bp = &sB[((t << 4) + lo16) * BPAD + (hi << 3)];
#pragma unroll
      for (int j = 0; j < 8; ++j) bf.u[j] = *(const uint32_t*)(bp + ((j < 4) ? (j << 1) : ((j << 1) + 8)));
      acc[t] = __builtin_amdgcn_wmma_f32_16x16x32_bf16(
          false, af.v, false, bf.v, (short)0, acc[t], false, false);
    }
    __syncthreads();
  }

  // ---- epilogue: C/D layout (VGPR v, lane) -> (row = v + 8*hi, col = lo16)
#pragma unroll
  for (int t = 0; t < 4; ++t) {
    int gn = n0 + (t << 4) + lo16;
    float bv = bias ? bias[gn] : 0.0f;
#pragma unroll
    for (int v = 0; v < 8; ++v) {
      int gm = m0 + (wave << 4) + v + (hi << 3);
      if (gm < M) {
        float o = acc[t][v] + bv;
        if (relu) o = fmaxf(o, 0.0f);
        Y[(size_t)gm * ldy + gn] = o;
      }
    }
  }
}

// ---------------------------------------------------------------------------
// Weight packing kernels (f32 -> bf16, transposed to [Nout][K])
__global__ void pack_bt_concat(const float* __restrict__ Wroot,
                               const float* __restrict__ Wrel,
                               uint16_t* __restrict__ BT,
                               int Kroot, int Ktot, int Nn) {
  int i = blockIdx.x * blockDim.x + threadIdx.x;
  if (i >= Nn * Ktot) return;
  int n = i / Ktot, k = i - n * Ktot;
  float v = (k < Kroot) ? Wroot[(size_t)k * Nn + n]
                        : Wrel[(size_t)(k - Kroot) * Nn + n];
  BT[i] = f32_to_bf16_rne(v);
}

__global__ void pack_qkvs(const float* __restrict__ Wq, const float* __restrict__ Wk,
                          const float* __restrict__ Wv, const float* __restrict__ Ws,
                          const float* __restrict__ bq, const float* __restrict__ bk,
                          const float* __restrict__ bv, const float* __restrict__ bs,
                          uint16_t* __restrict__ BT, float* __restrict__ biascat) {
  int i = blockIdx.x * blockDim.x + threadIdx.x;
  if (i >= 4 * HID * HID) return;
  int n = i >> 8, k = i & (HID - 1);
  int blk = n >> 8, nn = n & (HID - 1);
  const float* W = (blk == 0) ? Wq : (blk == 1) ? Wk : (blk == 2) ? Wv : Ws;
  BT[i] = f32_to_bf16_rne(W[(size_t)k * HID + nn]);
  if (k == 0) {
    const float* b = (blk == 0) ? bq : (blk == 1) ? bk : (blk == 2) ? bv : bs;
    biascat[n] = b[nn];
  }
}

__global__ void pack_cast(const float* __restrict__ W, uint16_t* __restrict__ BT, int count) {
  int i = blockIdx.x * blockDim.x + threadIdx.x;
  if (i < count) BT[i] = f32_to_bf16_rne(W[i]);
}

// ---------------------------------------------------------------------------
// Graph / elementwise kernels
__global__ void zero_f32(float* p, long long count) {
  long long i = (long long)blockIdx.x * blockDim.x + threadIdx.x;
  if (i < count) p[i] = 0.0f;
}

__global__ void init_xcat(const float* __restrict__ x, float* __restrict__ xcat, int N) {
  int i = blockIdx.x * blockDim.x + threadIdx.x;
  if (i >= N * 640) return;
  int n = i / 640, c = i - n * 640;
  xcat[i] = (c < F_IN) ? x[(size_t)n * F_IN + c] : 0.0f;
}

__global__ void zero_hcat_tail(float* __restrict__ hcat, int N) {
  int i = blockIdx.x * blockDim.x + threadIdx.x;
  if (i >= N * (HID * RELS)) return;
  int n = i >> 10, j = i & 1023;
  hcat[(size_t)n * 1280 + HID + j] = 0.0f;
}

__global__ void count_edges(const int* __restrict__ dst, const int* __restrict__ et,
                            float* __restrict__ cnt, int E) {
  int e = blockIdx.x * blockDim.x + threadIdx.x;
  if (e < E) atomicAdd(&cnt[(size_t)dst[e] * RELS + et[e]], 1.0f);
}

// scatter x[src] into xcat[dst, 128 + r*128 + :]
__global__ void agg_layer1(const int* __restrict__ src, const int* __restrict__ dst,
                           const int* __restrict__ et, const float* __restrict__ x,
                           float* __restrict__ xcat, int E) {
  int i = blockIdx.x * blockDim.x + threadIdx.x;
  if (i >= E * 32) return;
  int e = i >> 5, c = (i & 31) << 2;
  int s = src[e], d = dst[e], r = et[e];
  float4 v = *(const float4*)(x + (size_t)s * F_IN + c);
  float* o = xcat + (size_t)d * 640 + F_IN + r * F_IN + c;
  atomicAdd(o + 0, v.x); atomicAdd(o + 1, v.y);
  atomicAdd(o + 2, v.z); atomicAdd(o + 3, v.w);
}

// scatter h1 (= hcat cols 0..255) into hcat[dst, 256 + r*256 + :]
__global__ void agg_layer2(const int* __restrict__ src, const int* __restrict__ dst,
                           const int* __restrict__ et, const float* __restrict__ hcat_in,
                           float* __restrict__ hcat, int E) {
  int i = blockIdx.x * blockDim.x + threadIdx.x;
  if (i >= E * 64) return;
  int e = i >> 6, c = (i & 63) << 2;
  int s = src[e], d = dst[e], r = et[e];
  float4 v = *(const float4*)(hcat_in + (size_t)s * 1280 + c);
  float* o = hcat + (size_t)d * 1280 + HID + r * HID + c;
  atomicAdd(o + 0, v.x); atomicAdd(o + 1, v.y);
  atomicAdd(o + 2, v.z); atomicAdd(o + 3, v.w);
}

// divide aggregated sums by max(count,1) per (node, relation)
__global__ void norm_agg(float* __restrict__ agg, const float* __restrict__ cnt,
                         int N, int ld, int off, int F) {
  int i = blockIdx.x * blockDim.x + threadIdx.x;
  int width = RELS * F;
  if (i >= N * width) return;
  int n = i / width, j = i - n * width;
  int r = j / F;
  agg[(size_t)n * ld + off + j] *= 1.0f / fmaxf(cnt[(size_t)n * RELS + r], 1.0f);
}

// copy skip branch (qkvs cols 768..1023, bias already applied) into attn out
__global__ void attn_init(const float* __restrict__ qkvs, float* __restrict__ out, int N) {
  int i = blockIdx.x * blockDim.x + threadIdx.x;
  if (i >= N * HID) return;
  int n = i >> 8, c = i & 255;
  out[i] = qkvs[(size_t)n * 1024 + 768 + c];
}

__global__ void attn_score(const int* __restrict__ src, const int* __restrict__ dst,
                           const float* __restrict__ qkvs, float* __restrict__ score,
                           unsigned* __restrict__ smax, int E) {
  int i = blockIdx.x * blockDim.x + threadIdx.x;
  if (i >= E * HEADS) return;
  int e = i >> 2, h = i & 3;
  int s = src[e], d = dst[e];
  const float* q = qkvs + (size_t)d * 1024 + h * HEADC;         // q block
  const float* k = qkvs + (size_t)s * 1024 + HID + h * HEADC;   // k block
  float acc = 0.0f;
#pragma unroll
  for (int c = 0; c < HEADC; c += 4) {
    float4 qa = *(const float4*)(q + c);
    float4 ka = *(const float4*)(k + c);
    acc += qa.x * ka.x + qa.y * ka.y + qa.z * ka.z + qa.w * ka.w;
  }
  acc *= 0.125f;  // 1/sqrt(64)
  score[i] = acc;
  atomicMax(&smax[(size_t)d * HEADS + h], ford_enc(acc));
}

__global__ void attn_exp(const int* __restrict__ dst, const float* __restrict__ score,
                         const unsigned* __restrict__ smax, float* __restrict__ ex,
                         float* __restrict__ denom, int E) {
  int i = blockIdx.x * blockDim.x + threadIdx.x;
  if (i >= E * HEADS) return;
  int e = i >> 2, h = i & 3;
  int d = dst[e];
  float m = ford_dec(smax[(size_t)d * HEADS + h]);
  float v = expf(score[i] - m);
  ex[i] = v;
  atomicAdd(&denom[(size_t)d * HEADS + h], v);
}

__global__ void attn_scatter(const int* __restrict__ src, const int* __restrict__ dst,
                             const float* __restrict__ qkvs, const float* __restrict__ ex,
                             const float* __restrict__ denom, float* __restrict__ out, int E) {
  int i = blockIdx.x * blockDim.x + threadIdx.x;
  if (i >= E * 64) return;
  int e = i >> 6, c = (i & 63) << 2;
  int h = c >> 6;
  int s = src[e], d = dst[e];
  float alpha = ex[(size_t)e * HEADS + h] /
                fmaxf(denom[(size_t)d * HEADS + h], 1e-16f);
  float4 v = *(const float4*)(qkvs + (size_t)s * 1024 + 512 + c);  // v block
  float* o = out + (size_t)d * HID + c;
  atomicAdd(o + 0, alpha * v.x); atomicAdd(o + 1, alpha * v.y);
  atomicAdd(o + 2, alpha * v.z); atomicAdd(o + 3, alpha * v.w);
}

__global__ void gru_kernel(const float* __restrict__ gi, const float* __restrict__ gh,
                           const float* __restrict__ hprev, float* __restrict__ out, int N) {
  int i = blockIdx.x * blockDim.x + threadIdx.x;
  if (i >= N * HID) return;
  int n = i >> 8, c = i & 255;
  size_t b = (size_t)n * (3 * HID);
  float r  = sigmoidf_(gi[b + c] + gh[b + c]);
  float z  = sigmoidf_(gi[b + HID + c] + gh[b + HID + c]);
  float nw = tanhf(gi[b + 2 * HID + c] + r * gh[b + 2 * HID + c]);
  out[i] = (1.0f - z) * nw + z * hprev[i];
}

// ---------------------------------------------------------------------------
#define CDIV(a, b) (((a) + (b) - 1) / (b))

extern "C" void kernel_launch(void* const* d_in, const int* in_sizes, int n_in,
                              void* d_out, int out_size, void* d_ws, size_t ws_size,
                              hipStream_t stream) {
  const float* x      = (const float*)d_in[0];
  const int*   ei     = (const int*)d_in[1];
  const int*   et     = (const int*)d_in[2];
  const float* h_prev = (const float*)d_in[3];
  const float* W1     = (const float*)d_in[4];
  const float* root1  = (const float*)d_in[5];
  const float* b1     = (const float*)d_in[6];
  const float* W2     = (const float*)d_in[7];
  const float* root2  = (const float*)d_in[8];
  const float* b2     = (const float*)d_in[9];
  const float* Wq     = (const float*)d_in[10];
  const float* bq     = (const float*)d_in[11];
  const float* Wk     = (const float*)d_in[12];
  const float* bk     = (const float*)d_in[13];
  const float* Wv     = (const float*)d_in[14];
  const float* bv     = (const float*)d_in[15];
  const float* Wskip  = (const float*)d_in[16];
  const float* bskip  = (const float*)d_in[17];
  const float* W_ih   = (const float*)d_in[18];
  const float* b_ih   = (const float*)d_in[19];
  const float* W_hh   = (const float*)d_in[20];
  const float* b_hh   = (const float*)d_in[21];

  const int N = in_sizes[0] / F_IN;   // 20000
  const int E = in_sizes[2];          // 320000
  const int* src = ei;
  const int* dst = ei + E;
  float* out = (float*)d_out;

  // ---- workspace carve-out (offsets only; no allocation)
  char* ws = (char*)d_ws;
  size_t off = 0;
  auto wsalloc = [&](size_t bytes) -> void* {
    void* p = ws + off;
    off = (off + bytes + 255) & ~(size_t)255;
    return p;
  };
  float*    cnt   = (float*)wsalloc((size_t)N * RELS * 4);
  unsigned* smax  = (unsigned*)wsalloc((size_t)N * HEADS * 4);
  float*    denom = (float*)wsalloc((size_t)N * HEADS * 4);
  float*    xcat  = (float*)wsalloc((size_t)N * 640 * 4);    // [x | agg1]
  float*    hcat  = (float*)wsalloc((size_t)N * 1280 * 4);   // [h1 | agg2]
  float*    h2    = (float*)wsalloc((size_t)N * HID * 4);
  float*    qkvs  = (float*)wsalloc((size_t)N * 1024 * 4);
  float*    attn  = (float*)wsalloc((size_t)N * HID * 4);
  float*    score = (float*)wsalloc((size_t)E * HEADS * 4);
  float*    exs   = (float*)wsalloc((size_t)E * HEADS * 4);
  float*    gi    = (float*)wsalloc((size_t)N * 768 * 4);
  float*    gh    = (float*)wsalloc((size_t)N * 768 * 4);
  uint16_t* BT1   = (uint16_t*)wsalloc((size_t)HID * 640 * 2);
  uint16_t* BT2   = (uint16_t*)wsalloc((size_t)HID * 1280 * 2);
  uint16_t* BTq   = (uint16_t*)wsalloc((size_t)1024 * HID * 2);
  uint16_t* BTih  = (uint16_t*)wsalloc((size_t)768 * HID * 2);
  uint16_t* BThh  = (uint16_t*)wsalloc((size_t)768 * HID * 2);
  float*    bqkvs = (float*)wsalloc(1024 * 4);

  const int TB = 256;
  const int mgrid = CDIV(N, BM);

  // ---- pack weights (bf16, transposed to [Nout][K])
  pack_bt_concat<<<CDIV(HID * 640, TB), TB, 0, stream>>>(root1, W1, BT1, F_IN, 640, HID);
  pack_bt_concat<<<CDIV(HID * 1280, TB), TB, 0, stream>>>(root2, W2, BT2, HID, 1280, HID);
  pack_qkvs<<<CDIV(4 * HID * HID, TB), TB, 0, stream>>>(Wq, Wk, Wv, Wskip,
                                                        bq, bk, bv, bskip, BTq, bqkvs);
  pack_cast<<<CDIV(768 * HID, TB), TB, 0, stream>>>(W_ih, BTih, 768 * HID);
  pack_cast<<<CDIV(768 * HID, TB), TB, 0, stream>>>(W_hh, BThh, 768 * HID);

  // ---- init accumulators (cnt|smax|denom are contiguous: 12*N words)
  zero_f32<<<CDIV((long long)N * 12, TB), TB, 0, stream>>>(cnt, (long long)N * 12);
  init_xcat<<<CDIV(N * 640, TB), TB, 0, stream>>>(x, xcat, N);
  zero_hcat_tail<<<CDIV(N * 1024, TB), TB, 0, stream>>>(hcat, N);

  // ---- RGCN layer 1: aggregate-then-transform, one concat GEMM (K=640)
  count_edges<<<CDIV(E, TB), TB, 0, stream>>>(dst, et, cnt, E);
  agg_layer1<<<CDIV(E * 32, TB), TB, 0, stream>>>(src, dst, et, x, xcat, E);
  norm_agg<<<CDIV(N * RELS * F_IN, TB), TB, 0, stream>>>(xcat, cnt, N, 640, F_IN, F_IN);
  gemm_bf16_wmma<<<dim3(mgrid, HID / BN), TB, 0, stream>>>(
      xcat, 640, BT1, b1, hcat, 1280, N, 640, HID, 1);

  // ---- RGCN layer 2: same graph counts, one concat GEMM (K=1280)
  agg_layer2<<<CDIV(E * 64, TB), TB, 0, stream>>>(src, dst, et, hcat, hcat, E);
  norm_agg<<<CDIV(N * RELS * HID, TB), TB, 0, stream>>>(hcat, cnt, N, 1280, HID, HID);
  gemm_bf16_wmma<<<dim3(mgrid, HID / BN), TB, 0, stream>>>(
      hcat, 1280, BT2, b2, h2, HID, N, 1280, HID, 1);

  // ---- TransformerConv: fused q|k|v|skip GEMM (N=1024), then edge softmax
  gemm_bf16_wmma<<<dim3(mgrid, 1024 / BN), TB, 0, stream>>>(
      h2, HID, BTq, bqkvs, qkvs, 1024, N, HID, 1024, 0);
  attn_init<<<CDIV(N * HID, TB), TB, 0, stream>>>(qkvs, attn, N);
  attn_score<<<CDIV(E * HEADS, TB), TB, 0, stream>>>(src, dst, qkvs, score, smax, E);
  attn_exp<<<CDIV(E * HEADS, TB), TB, 0, stream>>>(dst, score, smax, exs, denom, E);
  attn_scatter<<<CDIV(E * 64, TB), TB, 0, stream>>>(src, dst, qkvs, exs, denom, attn, E);

  // ---- GRU: two GEMMs (W stored [3H,H] row-major == already [Nout][K]) + fuse
  gemm_bf16_wmma<<<dim3(mgrid, 768 / BN), TB, 0, stream>>>(
      attn, HID, BTih, b_ih, gi, 768, N, HID, 768, 0);
  gemm_bf16_wmma<<<dim3(mgrid, 768 / BN), TB, 0, stream>>>(
      h_prev, HID, BThh, b_hh, gh, 768, N, HID, 768, 0);
  gru_kernel<<<CDIV(N * HID, TB), TB, 0, stream>>>(gi, gh, h_prev, out, N);
}
